// MemoryEfficientAttention_16690242912687
// MI455X (gfx1250) — matrix operand order repaired and checked by reading, not compiled, not run on hardware
//
#include <hip/hip_runtime.h>
#include <hip/hip_bf16.h>

// ---------------------------------------------------------------------------
// Strided-sparse attention block for MI455X (gfx1250), bf16 WMMA throughout.
//   B=2, S=2048, H=1024, NH=16, D=64
//   - qkv_proj : x@{q,k,v}_w -> bf16 [B,NH,S,D] ws (Q scaled by sigmoid(phi))
//   - sparse_attn: 4 waves/block share the strided (j%4==0) K/V staging;
//                  strided K tile fetched by the Tensor Data Mover (row stride
//                  4*64 elements, LDS row padding via descriptor); Q + local
//                  windows staged via async global->LDS; flash softmax.
//   - out_proj : attn@o_w + o_b -> f32 d_out
// Workspace: Q,K,V,attn bf16 = 4 * 8MB = 32MB.
// ---------------------------------------------------------------------------

#define Hdim 1024
#define SEQ  2048
#define NHEAD 16
#define DH   64

typedef __attribute__((ext_vector_type(16))) __bf16 bf16x16;
typedef __attribute__((ext_vector_type(8)))  __bf16 bf16x8;
typedef __attribute__((ext_vector_type(8)))  float  v8f;
typedef int v4i           __attribute__((vector_size(16)));
typedef int v8i           __attribute__((vector_size(32)));
typedef unsigned int v4ui __attribute__((vector_size(16)));

#if defined(__has_builtin)
#if __has_builtin(__builtin_amdgcn_global_load_async_to_lds_b128) && \
    __has_builtin(__builtin_amdgcn_s_wait_asynccnt)
#define USE_ASYNC 1
#endif
#if __has_builtin(__builtin_amdgcn_tensor_load_to_lds) && \
    __has_builtin(__builtin_amdgcn_s_wait_tensorcnt)
#define USE_TDM 1
#endif
#endif

// 16-byte global->LDS copy: async-DMA path on CDNA5, sync fallback otherwise.
__device__ __forceinline__ void cp16(unsigned short* l, const unsigned short* g) {
#ifdef USE_ASYNC
  __builtin_amdgcn_global_load_async_to_lds_b128(
      (__attribute__((address_space(1))) v4i*)(g),
      (__attribute__((address_space(3))) v4i*)(l), 0, 0);
#else
  *(uint4*)l = *(const uint4*)g;
#endif
}
__device__ __forceinline__ void async_wait() {
#ifdef USE_ASYNC
  __builtin_amdgcn_s_wait_asynccnt(0);
#endif
}

#ifdef USE_TDM
// TDM: load a (rows x 64 bf16) tile, row stride `stride` elements, into LDS
// rows padded to 72 shorts (pad_interval=32 DWORDs, pad_amount=4 DWORDs).
__device__ __forceinline__ void tdm_load_tile(const unsigned short* gsrc,
                                              unsigned lds_off,
                                              unsigned rows, unsigned stride) {
  unsigned long long ga = (unsigned long long)(__UINTPTR_TYPE__)gsrc;
  v4ui g0 = { 1u,                                 // count=1, user descriptor
              lds_off,                            // lds_addr (bytes)
              (unsigned)ga,                       // global_addr[31:0]
              ((unsigned)(ga >> 32) & 0x01FFFFFFu) | 0x80000000u }; // addr[56:32] | type=2
  v8i g1 = { (int)0x07110000u,                    // data_size=2B, pad_en, int=32dw, amt=4dw
             (int)(64u << 16),                    // tensor_dim0 = 64
             (int)((unsigned)SEQ << 16),          // tensor_dim1
             (int)(64u << 16),                    // tile_dim0 = 64
             (int)rows,                           // tile_dim1 = rows
             (int)stride,                         // tensor_dim0_stride (elements)
             0, 0 };
  v4i z4 = { 0, 0, 0, 0 };
#if __clang_major__ >= 23
  v8i z8 = { 0, 0, 0, 0, 0, 0, 0, 0 };
  __builtin_amdgcn_tensor_load_to_lds(g0, g1, z4, z4, z8, 0);
#else
  __builtin_amdgcn_tensor_load_to_lds(g0, g1, z4, z4, 0);
#endif
}
__device__ __forceinline__ unsigned lds_offset_of(const unsigned short* p) {
  return (unsigned)(unsigned long long)
      (__attribute__((address_space(3))) unsigned short*)p;
}
#endif

__device__ __forceinline__ unsigned short f2bf(float f) {
  unsigned int u = __float_as_uint(f);
  u += 0x7FFFu + ((u >> 16) & 1u);        // round-to-nearest-even
  return (unsigned short)(u >> 16);
}

__device__ __forceinline__ bf16x16 frag16(const unsigned short* lo,
                                          const unsigned short* hi) {
  bf16x8 a = *(const bf16x8*)lo;           // ds_load_b128
  bf16x8 b = *(const bf16x8*)hi;           // ds_load_b128
  return __builtin_shufflevector(a, b, 0,1,2,3,4,5,6,7,8,9,10,11,12,13,14,15);
}

__device__ __forceinline__ v8f wmma_bf16(bf16x16 a, bf16x16 b, v8f c) {
  return __builtin_amdgcn_wmma_f32_16x16x32_bf16(
      false, a, false, b, (short)0, c, false, false);
}

__device__ __forceinline__ float rmax16(float v) {
  for (int m = 1; m < 16; m <<= 1) v = fmaxf(v, __shfl_xor(v, m, 32));
  return v;
}
__device__ __forceinline__ float rsum16(float v) {
  for (int m = 1; m < 16; m <<= 1) v += __shfl_xor(v, m, 32);
  return v;
}

// ---------------------------------------------------------------------------
// Kernel 1: QKV projection.  grid (N/64, M/32, 3), block 256 (8 waves).
// ---------------------------------------------------------------------------
__global__ __launch_bounds__(256)
void qkv_proj(const float* __restrict__ x,
              const float* __restrict__ qw, const float* __restrict__ kw,
              const float* __restrict__ vw, const float* __restrict__ uscale,
              unsigned short* __restrict__ oq, unsigned short* __restrict__ ok,
              unsigned short* __restrict__ ov) {
  __shared__ __align__(16) unsigned short lA[32][40];  // [m][k]
  __shared__ __align__(16) unsigned short lB[64][40];  // [n][k] (weight transposed)

  const int tid  = threadIdx.x;
  const int lane = tid & 31, wave = tid >> 5;
  const int half = lane >> 4, ln = lane & 15;
  const int wm = wave >> 2, wn = wave & 3;
  const int M0 = blockIdx.y * 32, N0 = blockIdx.x * 64;
  const int z  = blockIdx.z;
  const float* w = (z == 0) ? qw : (z == 1) ? kw : vw;
  unsigned short* dst = (z == 0) ? oq : (z == 1) ? ok : ov;

  v8f acc = {};
  for (int k0 = 0; k0 < Hdim; k0 += 32) {
    if (k0 + 32 < Hdim) {                               // prefetch next tiles
      __builtin_prefetch(&x[(size_t)(M0 + (tid >> 3)) * Hdim + k0 + 32 + (tid & 7) * 4], 0, 3);
      __builtin_prefetch(&w[(size_t)(k0 + 32 + (tid >> 6)) * Hdim + N0 + (tid & 63)], 0, 3);
    }
    for (int e = 0; e < 4; ++e) {                       // 32x32 A tile, f32->bf16
      int idx = e * 256 + tid;
      int m = idx >> 5, k = idx & 31;
      lA[m][k] = f2bf(x[(size_t)(M0 + m) * Hdim + k0 + k]);
    }
    for (int e = 0; e < 8; ++e) {                       // 32k x 64n B tile, transposed
      int idx = e * 256 + tid;
      int k = idx >> 6, n = idx & 63;
      lB[n][k] = f2bf(w[(size_t)(k0 + k) * Hdim + N0 + n]);
    }
    __syncthreads();
    bf16x16 a = frag16(&lA[16 * wm + ln][16 * half],
                       &lA[16 * wm + ln][16 * half + 8]);
    bf16x16 b = frag16(&lB[16 * wn + ln][16 * half],
                       &lB[16 * wn + ln][16 * half + 8]);
    acc = wmma_bf16(a, b, acc);
    __syncthreads();
  }

  float scl = 1.0f;
  if (z == 0) {                                          // sigmoid(unity_scale)
    float s = uscale[0];
    scl = __builtin_amdgcn_rcpf(1.0f + __expf(-s));
  }
  for (int r = 0; r < 8; ++r) {
    int gm = M0 + 16 * wm + r + 8 * half;
    int gn = N0 + 16 * wn + ln;
    int b_ = gm >> 11, s_ = gm & (SEQ - 1);
    int nh = gn >> 6, d = gn & 63;
    dst[(((size_t)b_ * NHEAD + nh) * SEQ + s_) * DH + d] = f2bf(acc[r] * scl);
  }
}

// ---------------------------------------------------------------------------
// Kernel 2: strided-sparse flash attention.
// grid (S/64, B*NH), block 128 (4 waves; wave w owns query tile qb+16w).
// Strided K tile: one TDM descriptor per iteration (row stride 4*64 elems).
// Q + local windows: async global->LDS.  V: manual transposing scatter.
// ---------------------------------------------------------------------------
__global__ __launch_bounds__(128)
void sparse_attn(const unsigned short* __restrict__ Q,
                 const unsigned short* __restrict__ K,
                 const unsigned short* __restrict__ V,
                 unsigned short* __restrict__ O) {
  __shared__ __align__(16) unsigned short lQ[64][72];    // [row][d]
  __shared__ __align__(16) unsigned short lK[32][72];    // strided keys [t][d]
  __shared__ __align__(16) unsigned short lVT[64][40];   // strided V^T [d][t]
  __shared__ __align__(16) unsigned short lKl[80][72];   // local window keys
  __shared__ __align__(16) unsigned short lVTl[64][88];  // local window V^T
  __shared__ __align__(16) unsigned short lP[4][16][40]; // per-wave probs

  const int tid  = threadIdx.x;
  const int lane = tid & 31, wv = tid >> 5;
  const int half = lane >> 4, ln = lane & 15;
  const int qb = blockIdx.x * 64;
  const int q0 = qb + 16 * wv;
  const int bh = blockIdx.y;                             // b*NHEAD + h
  const size_t base = (size_t)bh * SEQ * DH;
  const unsigned short* Qb = Q + base;
  const unsigned short* Kb = K + base;
  const unsigned short* Vb = V + base;

  // stage Q (64 x 64) -- async 16B copies
  for (int e = 0; e < 4; ++e) {
    int idx = e * 128 + tid;                             // 512 chunks
    int r = idx >> 3, c = (idx & 7) * 8;
    cp16(&lQ[r][c], Qb + (size_t)(qb + r) * DH + c);
  }
  // stage local windows once: 80 keys j = qb-8 .. qb+71 (clamped)
  for (int e = 0; e < 5; ++e) {
    int idx = e * 128 + tid;                             // 640 chunks
    int tg = idx >> 3, c = (idx & 7) * 8;
    int j = qb - 8 + tg;
    j = j < 0 ? 0 : (j > SEQ - 1 ? SEQ - 1 : j);
    cp16(&lKl[tg][c], Kb + (size_t)j * DH + c);
    uint4 vv = *(const uint4*)(Vb + (size_t)j * DH + c);
    unsigned int wd[4] = {vv.x, vv.y, vv.z, vv.w};
    for (int q = 0; q < 4; ++q) {
      lVTl[c + 2 * q][tg]     = (unsigned short)(wd[q] & 0xffffu);
      lVTl[c + 2 * q + 1][tg] = (unsigned short)(wd[q] >> 16);
    }
  }
  async_wait();
  __syncthreads();

  bf16x16 aq0 = frag16(&lQ[16 * wv + ln][16 * half],
                       &lQ[16 * wv + ln][16 * half + 8]);          // d 0..31
  bf16x16 aq1 = frag16(&lQ[16 * wv + ln][32 + 16 * half],
                       &lQ[16 * wv + ln][32 + 16 * half + 8]);     // d 32..63

  v8f o0 = {}, o1 = {}, o2 = {}, o3 = {};
  float mrow[8], lrow[8];
  for (int r = 0; r < 8; ++r) { mrow[r] = -1e30f; lrow[r] = 0.0f; }

  // ------------------ strided part: dense over 512 compacted keys ----------
  for (int it = 0; it < 16; ++it) {
#ifdef USE_TDM
    if (wv == 0)                                         // one DMA for 32 key rows
      tdm_load_tile(Kb + (size_t)(128 * it) * DH, lds_offset_of(&lK[0][0]),
                    32u, 4u * DH);
    for (int e = 0; e < 2; ++e) {                        // V: transposing scatter
      int idx = e * 128 + tid;
      int t = idx >> 3, c = (idx & 7) * 8;
      int j = 128 * it + 4 * t;
      uint4 vv = *(const uint4*)(Vb + (size_t)j * DH + c);
      unsigned int wd[4] = {vv.x, vv.y, vv.z, vv.w};
      for (int q = 0; q < 4; ++q) {
        lVT[c + 2 * q][t]     = (unsigned short)(wd[q] & 0xffffu);
        lVT[c + 2 * q + 1][t] = (unsigned short)(wd[q] >> 16);
      }
    }
    if (wv == 0) __builtin_amdgcn_s_wait_tensorcnt(0);
#else
    for (int e = 0; e < 2; ++e) {                        // 256 chunks, shared
      int idx = e * 128 + tid;
      int t = idx >> 3, c = (idx & 7) * 8;
      int j = 128 * it + 4 * t;                          // compacted strided key
      cp16(&lK[t][c], Kb + (size_t)j * DH + c);
      uint4 vv = *(const uint4*)(Vb + (size_t)j * DH + c);
      unsigned int wd[4] = {vv.x, vv.y, vv.z, vv.w};
      for (int q = 0; q < 4; ++q) {
        lVT[c + 2 * q][t]     = (unsigned short)(wd[q] & 0xffffu);
        lVT[c + 2 * q + 1][t] = (unsigned short)(wd[q] >> 16);
      }
    }
    async_wait();
#endif
    __syncthreads();

    v8f s0 = {}, s1 = {};
    {
      bf16x16 b00 = frag16(&lK[ln][16 * half],      &lK[ln][16 * half + 8]);
      bf16x16 b01 = frag16(&lK[ln][32 + 16 * half], &lK[ln][32 + 16 * half + 8]);
      s0 = wmma_bf16(aq0, b00, s0);
      s0 = wmma_bf16(aq1, b01, s0);
      bf16x16 b10 = frag16(&lK[16 + ln][16 * half],      &lK[16 + ln][16 * half + 8]);
      bf16x16 b11 = frag16(&lK[16 + ln][32 + 16 * half], &lK[16 + ln][32 + 16 * half + 8]);
      s1 = wmma_bf16(aq0, b10, s1);
      s1 = wmma_bf16(aq1, b11, s1);
    }

    for (int r = 0; r < 8; ++r) {                        // online softmax
      float sc0 = s0[r] * 0.125f, sc1 = s1[r] * 0.125f;
      float mt = rmax16(fmaxf(sc0, sc1));
      float mn = fmaxf(mrow[r], mt);
      float al = __expf(mrow[r] - mn);
      mrow[r]  = mn;
      float p0 = __expf(sc0 - mn);
      float p1 = __expf(sc1 - mn);
      lrow[r]  = lrow[r] * al + rsum16(p0 + p1);
      lP[wv][r + 8 * half][ln]      = f2bf(p0);
      lP[wv][r + 8 * half][16 + ln] = f2bf(p1);
      o0[r] *= al; o1[r] *= al; o2[r] *= al; o3[r] *= al;
    }
    // O += P(16x32) * V(32x64)  (lP is per-wave private: no barrier needed)
    bf16x16 ap = frag16(&lP[wv][ln][16 * half], &lP[wv][ln][16 * half + 8]);
    o0 = wmma_bf16(ap, frag16(&lVT[ln][16 * half],      &lVT[ln][16 * half + 8]), o0);
    o1 = wmma_bf16(ap, frag16(&lVT[16 + ln][16 * half], &lVT[16 + ln][16 * half + 8]), o1);
    o2 = wmma_bf16(ap, frag16(&lVT[32 + ln][16 * half], &lVT[32 + ln][16 * half + 8]), o2);
    o3 = wmma_bf16(ap, frag16(&lVT[48 + ln][16 * half], &lVT[48 + ln][16 * half + 8]), o3);
    __syncthreads();
  }

  // ------------------ local window pass (masked, per wave) -----------------
  {
    const int r0 = 16 * wv;                              // this wave's window slice
    v8f s0 = {}, s1 = {};
    bf16x16 b00 = frag16(&lKl[r0 + ln][16 * half],      &lKl[r0 + ln][16 * half + 8]);
    bf16x16 b01 = frag16(&lKl[r0 + ln][32 + 16 * half], &lKl[r0 + ln][32 + 16 * half + 8]);
    s0 = wmma_bf16(aq0, b00, s0);
    s0 = wmma_bf16(aq1, b01, s0);
    bf16x16 b10 = frag16(&lKl[r0 + 16 + ln][16 * half],      &lKl[r0 + 16 + ln][16 * half + 8]);
    bf16x16 b11 = frag16(&lKl[r0 + 16 + ln][32 + 16 * half], &lKl[r0 + 16 + ln][32 + 16 * half + 8]);
    s1 = wmma_bf16(aq0, b10, s1);
    s1 = wmma_bf16(aq1, b11, s1);

    for (int r = 0; r < 8; ++r) {
      float sc0 = s0[r] * 0.125f, sc1 = s1[r] * 0.125f;
      int i  = q0 + r + 8 * half;
      int j0 = q0 - 8 + ln;                              // subtile 0 column
      int j1 = j0 + 16;                                  // subtile 1 column
      int d0 = i - j0; if (d0 < 0) d0 = -d0;
      int d1 = i - j1; if (d1 < 0) d1 = -d1;
      if (!((j0 >= 0) && (j0 < SEQ) && (d0 <= 8) && ((j0 & 3) != 0))) sc0 = -1e30f;
      if (!((j1 >= 0) && (j1 < SEQ) && (d1 <= 8) && ((j1 & 3) != 0))) sc1 = -1e30f;
      float mt = rmax16(fmaxf(sc0, sc1));
      float mn = fmaxf(mrow[r], mt);
      float al = __expf(mrow[r] - mn);
      mrow[r]  = mn;
      float p0 = __expf(sc0 - mn);
      float p1 = __expf(sc1 - mn);
      lrow[r]  = lrow[r] * al + rsum16(p0 + p1);
      lP[wv][r + 8 * half][ln]      = f2bf(p0);
      lP[wv][r + 8 * half][16 + ln] = f2bf(p1);
      o0[r] *= al; o1[r] *= al; o2[r] *= al; o3[r] *= al;
    }
    bf16x16 ap = frag16(&lP[wv][ln][16 * half], &lP[wv][ln][16 * half + 8]);
    o0 = wmma_bf16(ap, frag16(&lVTl[ln][r0 + 16 * half],      &lVTl[ln][r0 + 16 * half + 8]), o0);
    o1 = wmma_bf16(ap, frag16(&lVTl[16 + ln][r0 + 16 * half], &lVTl[16 + ln][r0 + 16 * half + 8]), o1);
    o2 = wmma_bf16(ap, frag16(&lVTl[32 + ln][r0 + 16 * half], &lVTl[32 + ln][r0 + 16 * half + 8]), o2);
    o3 = wmma_bf16(ap, frag16(&lVTl[48 + ln][r0 + 16 * half], &lVTl[48 + ln][r0 + 16 * half + 8]), o3);
  }

  // epilogue: normalize and store as bf16 [B,S,H]
  const int b_ = bh >> 4, h_ = bh & 15;
  for (int r = 0; r < 8; ++r) {
    float inv = __builtin_amdgcn_rcpf(lrow[r]);
    int s_ = q0 + r + 8 * half;
    size_t rowbase = ((size_t)b_ * SEQ + s_) * Hdim + (size_t)h_ * DH;
    O[rowbase + ln]      = f2bf(o0[r] * inv);
    O[rowbase + 16 + ln] = f2bf(o1[r] * inv);
    O[rowbase + 32 + ln] = f2bf(o2[r] * inv);
    O[rowbase + 48 + ln] = f2bf(o3[r] * inv);
  }
}

// ---------------------------------------------------------------------------
// Kernel 3: output projection (bf16 A from ws, async-staged) + bias -> f32.
// ---------------------------------------------------------------------------
__global__ __launch_bounds__(256)
void out_proj(const unsigned short* __restrict__ a,
              const float* __restrict__ w, const float* __restrict__ bias,
              float* __restrict__ out) {
  __shared__ __align__(16) unsigned short lA[32][40];
  __shared__ __align__(16) unsigned short lB[64][40];

  const int tid  = threadIdx.x;
  const int lane = tid & 31, wave = tid >> 5;
  const int half = lane >> 4, ln = lane & 15;
  const int wm = wave >> 2, wn = wave & 3;
  const int M0 = blockIdx.y * 32, N0 = blockIdx.x * 64;

  v8f acc = {};
  for (int k0 = 0; k0 < Hdim; k0 += 32) {
    if (k0 + 32 < Hdim)
      __builtin_prefetch(&w[(size_t)(k0 + 32 + (tid >> 6)) * Hdim + N0 + (tid & 63)], 0, 3);
    if (tid < 128) {                                     // 32 rows x 4 chunks async
      int m = tid >> 2, k = (tid & 3) * 8;
      cp16(&lA[m][k], a + (size_t)(M0 + m) * Hdim + k0 + k);
    }
    for (int e = 0; e < 8; ++e) {
      int idx = e * 256 + tid;
      int k = idx >> 6, n = idx & 63;
      lB[n][k] = f2bf(w[(size_t)(k0 + k) * Hdim + N0 + n]);
    }
    async_wait();
    __syncthreads();
    bf16x16 av = frag16(&lA[16 * wm + ln][16 * half],
                        &lA[16 * wm + ln][16 * half + 8]);
    bf16x16 bv = frag16(&lB[16 * wn + ln][16 * half],
                        &lB[16 * wn + ln][16 * half + 8]);
    acc = wmma_bf16(av, bv, acc);
    __syncthreads();
  }

  for (int r = 0; r < 8; ++r) {
    int gm = M0 + 16 * wm + r + 8 * half;
    int gn = N0 + 16 * wn + ln;
    out[(size_t)gm * Hdim + gn] = acc[r] + bias[gn];
  }
}

// ---------------------------------------------------------------------------
extern "C" void kernel_launch(void* const* d_in, const int* in_sizes, int n_in,
                              void* d_out, int out_size, void* d_ws, size_t ws_size,
                              hipStream_t stream) {
  const float* x  = (const float*)d_in[0];
  const float* qw = (const float*)d_in[1];
  const float* kw = (const float*)d_in[2];
  const float* vw = (const float*)d_in[3];
  const float* ow = (const float*)d_in[4];
  const float* ob = (const float*)d_in[5];
  const float* us = (const float*)d_in[6];
  float* out = (float*)d_out;

  const size_t per = (size_t)2 * NHEAD * SEQ * DH;       // 4M bf16 elements = 8MB
  unsigned short* wsq = (unsigned short*)d_ws;
  unsigned short* wsk = wsq + per;
  unsigned short* wsv = wsk + per;
  unsigned short* wsa = wsv + per;                       // attn out, [B,S,H] bf16

  qkv_proj<<<dim3(Hdim / 64, (2 * SEQ) / 32, 3), 256, 0, stream>>>(
      x, qw, kw, vw, us, wsq, wsk, wsv);

  sparse_attn<<<dim3(SEQ / 64, 2 * NHEAD), 128, 0, stream>>>(wsq, wsk, wsv, wsa);

  out_proj<<<dim3(Hdim / 64, (2 * SEQ) / 32), 256, 0, stream>>>(wsa, ow, ob, out);
}
